// OneNetSeqHead_26259430048563
// MI455X (gfx1250) — compile-verified, matrix operand order checked by reading, not run on hardware
//
#include <hip/hip_runtime.h>
#include <hip/hip_bf16.h>
#include <stdint.h>

typedef __attribute__((ext_vector_type(16))) _Float16 v16h;
typedef __attribute__((ext_vector_type(8)))  float    v8f;

#define HH 180
#define WW 180
#define HWP (HH*WW)
#define NB 4

// conv tiling: 32 output pixels per workgroup row-segment, 2 WMMA N-subtiles/wave
#define PTILE 32
#define PCOLS 34                  // halo: PTILE + 2
#define NCELLS (3 * PCOLS)        // 102
#define PATCH_H (NCELLS * 32)     // halves
#define PATCH_U (PATCH_H / 2)     // u32s

union H2 { _Float16 h[2]; uint32_t u; };

// ---------------------------------------------------------------------------
// Weight repack: OIHW f32 -> [pos][O][I] f16 (A-fragment rows K-contiguous)
// ---------------------------------------------------------------------------
__global__ void prep_w_k(const float* __restrict__ src, _Float16* __restrict__ dst,
                         int O, int I) {
  int idx = blockIdx.x * blockDim.x + threadIdx.x;
  int total = O * I * 9;
  if (idx >= total) return;
  int i = idx % I;
  int t = idx / I;
  int o = t % O;
  int pos = t / O;
  dst[idx] = (_Float16)src[(size_t)(o * I + i) * 9 + pos];
}

// BN fold: s = g*rsqrt(v+eps), t = b - m*s
__global__ void prep_bn_k(const float* __restrict__ g, const float* __restrict__ b,
                          const float* __restrict__ m, const float* __restrict__ v,
                          float* __restrict__ s, float* __restrict__ t, int C) {
  int c = blockIdx.x * blockDim.x + threadIdx.x;
  if (c < C) {
    float sv = g[c] * rsqrtf(v[c] + 1e-5f);
    s[c] = sv;
    t[c] = b[c] - m[c] * sv;
  }
}

// ---------------------------------------------------------------------------
// Implicit-GEMM 3x3 SAME conv via V_WMMA_F32_16X16X32_F16.
// Block = 128 threads (4 waves). Wave w computes out-channels [16w,16w+16)
// for 32 consecutive pixels of one row (2 N-subtiles sharing one A fragment).
// fmt   : input 0 = f32 NCHW, 1 = packed f16 channel-pair planes (u32)
// outFmt: output 0 = f32 strided into (outCstride,outCoff), 1 = packed f16 pairs
// postop: 0 = bias only, 1 = bias+(bn)+relu, 2 = bias+clip(-5,5)
// ---------------------------------------------------------------------------
__global__ __launch_bounds__(128)
void conv3x3_wmma(const void* __restrict__ in0, int C0,
                  const void* __restrict__ in1, int C1,
                  const _Float16* __restrict__ w16,
                  const float* __restrict__ bias,
                  const float* __restrict__ bnS, const float* __restrict__ bnT,
                  void* __restrict__ outv, int outCstride, int outCoff,
                  int Cout, int postop, int fmt, int outFmt) {
  // patch layout: [cell = row*34+col][k(32)] halves -> K pairs u32-contiguous
  __shared__ __align__(16) _Float16 patch[PATCH_H];
  uint32_t* patch32 = (uint32_t*)patch;

  const int tid  = threadIdx.x;
  const int lane = tid & 31;
  const int wave = tid >> 5;
  const int x0 = blockIdx.x * PTILE;
  const int y  = blockIdx.y;
  const int nb = blockIdx.z;
  const int Ctot = C0 + C1;
  const int co0 = wave * 16;
  const int mrow = lane & 15;  // A row / B col / D col
  const int hi   = lane >> 4;  // K-half select per WMMA 16-bit layout

  // per-thread fill cell (packed-input path): cell == tid
  const int frow = tid / PCOLS;
  const int fcol = tid - frow * PCOLS;
  const int fgy = y + frow - 1;
  const int fgx = x0 + fcol - 1;
  const bool cellOk = tid < NCELLS;
  const bool inb = cellOk && fgy >= 0 && fgy < HH && fgx >= 0 && fgx < WW;
  const uint32_t fpix4 = (uint32_t)(fgy * WW + fgx) * 4u;  // byte offset in plane
  const uint32_t lbase = (uint32_t)(uintptr_t)&patch32[tid * 16];

  v8f acc0 = {}, acc1 = {};
  const int nChunks = Ctot >> 5;

  const int co   = co0 + mrow;
  const bool cok = (co < Cout);
  const int cosel = cok ? co : 0;

  for (int ch = 0; ch < nChunks; ++ch) {
    const int ci0 = ch * 32;
    const void* srcv;
    int cbase, Csel;
    if (ci0 < C0) { srcv = in0; cbase = ci0;      Csel = C0; }
    else          { srcv = in1; cbase = ci0 - C0; Csel = C1; }

    __syncthreads();
    if (fmt == 0) {
      // f32 NCHW input: load + convert + pack pairs into LDS
      const float* src = (const float*)srcv;
      for (int e = tid; e < PATCH_U; e += 128) {
        int kpair = e & 15;          // k = 2*kpair
        int cell  = e >> 4;          // row*34 + col
        int col = cell % PCOLS;
        int row = cell / PCOLS;
        int gy = y + row - 1;
        int gx = x0 + col - 1;
        float a0 = 0.0f, a1 = 0.0f;
        if (gy >= 0 && gy < HH && gx >= 0 && gx < WW) {
          const float* p = src + ((size_t)(nb * Csel + cbase + kpair * 2)) * HWP +
                           gy * WW + gx;
          a0 = p[0];
          a1 = p[HWP];
        }
        H2 pk;
        pk.h[0] = (_Float16)a0;
        pk.h[1] = (_Float16)a1;
        patch32[e] = pk.u;
      }
    } else {
      // packed f16-pair input: raw u32 async copies global -> LDS
      const uint32_t* srcp = (const uint32_t*)srcv;
      const size_t planeBase = (size_t)(nb * (Csel >> 1) + (cbase >> 1)) * HWP;
#pragma unroll 4
      for (int kp = 0; kp < 16; ++kp) {
        const uint32_t* sb = srcp + planeBase + (size_t)kp * HWP;
        if (inb) {
          uint32_t la = lbase + (uint32_t)kp * 4u;
          asm volatile("global_load_async_to_lds_b32 %0, %1, %2"
                       :: "v"(la), "v"(fpix4), "s"(sb)
                       : "memory");
        } else if (cellOk) {
          patch32[tid * 16 + kp] = 0u;
        }
      }
      asm volatile("s_wait_asynccnt 0x0" ::: "memory");
    }
    __syncthreads();

#pragma unroll
    for (int pos = 0; pos < 9; ++pos) {
      const int ky = pos / 3;
      const int kx = pos % 3;
      union { uint4 q[2]; v16h h; } A, B0, B1;
      // A fragment: two 128-bit loads selected by K-half (lanes 0-15 vs 16-31)
      const uint4* w4 =
          (const uint4*)(w16 + ((size_t)(pos * Cout + cosel)) * Ctot + ci0);
      if (cok) {
        A.q[0] = w4[hi];
        A.q[1] = w4[2 + hi];
      } else {
        uint4 z; z.x = 0; z.y = 0; z.z = 0; z.w = 0;
        A.q[0] = z; A.q[1] = z;
      }
      // B fragments from LDS (64B-aligned cell base -> ds_load_b128)
      const uint4* b4a =
          (const uint4*)&patch[((ky * PCOLS) + kx + mrow) * 32];
      const uint4* b4b =
          (const uint4*)&patch[((ky * PCOLS) + kx + 16 + mrow) * 32];
      B0.q[0] = b4a[hi];
      B0.q[1] = b4a[2 + hi];
      B1.q[0] = b4b[hi];
      B1.q[1] = b4b[2 + hi];
      acc0 = __builtin_amdgcn_wmma_f32_16x16x32_f16(
          false, A.h, false, B0.h, (short)0, acc0, false, false);
      acc1 = __builtin_amdgcn_wmma_f32_16x16x32_f16(
          false, A.h, false, B1.h, (short)0, acc1, false, false);
    }
  }

  // epilogue: D layout -> VGPR r holds row (r + 8*hi), col = lane&15
  const int cbl = co0 + hi * 8;  // this lane holds channels cbl..cbl+7
#pragma unroll
  for (int sub = 0; sub < 2; ++sub) {
    const int x = x0 + sub * 16 + mrow;
    if (x >= WW) continue;
    if (outFmt == 1) {
      // packed f16 channel-pair output (Cout==64 layers)
      uint32_t* outp = (uint32_t*)outv;
#pragma unroll
      for (int j = 0; j < 4; ++j) {
        float v0 = (sub == 0 ? acc0[2 * j]     : acc1[2 * j])     + bias[cbl + 2 * j];
        float v1 = (sub == 0 ? acc0[2 * j + 1] : acc1[2 * j + 1]) + bias[cbl + 2 * j + 1];
        if (bnS) {
          v0 = v0 * bnS[cbl + 2 * j]     + bnT[cbl + 2 * j];
          v1 = v1 * bnS[cbl + 2 * j + 1] + bnT[cbl + 2 * j + 1];
        }
        if (postop == 1) { v0 = fmaxf(v0, 0.0f); v1 = fmaxf(v1, 0.0f); }
        H2 pk;
        pk.h[0] = (_Float16)v0;
        pk.h[1] = (_Float16)v1;
        outp[((size_t)(nb * (Cout >> 1) + (cbl >> 1) + j)) * HWP + y * WW + x] = pk.u;
      }
    } else {
      float* outp = (float*)outv;
#pragma unroll
      for (int r = 0; r < 8; ++r) {
        int coz = cbl + r;
        if (coz < Cout) {
          float vv = (sub == 0 ? acc0[r] : acc1[r]) + bias[coz];
          if (bnS) vv = vv * bnS[coz] + bnT[coz];
          if (postop == 1)      vv = fmaxf(vv, 0.0f);
          else if (postop == 2) vv = fminf(fmaxf(vv, -5.0f), 5.0f);
          outp[((size_t)(nb * outCstride + outCoff + coz)) * HWP + y * WW + x] = vv;
        }
      }
    }
  }
}

// ---------------------------------------------------------------------------
// Bilinear sampling helpers (grid_sample, zero pad, align_corners=False)
// ---------------------------------------------------------------------------
__device__ __forceinline__ float tgrid(float r) {
  // ((raw - a)/xall*2 - 1 + 1)*(W/2) - 0.5 with a=-53.7, xall=107.4, W=H=180
  return (r + 53.7f) * (180.0f / 107.4f) - 0.5f;
}

__device__ __forceinline__ void bilin_setup(float u, float v,
                                            int& i00, int& i01, int& i10, int& i11,
                                            float& w00, float& w01, float& w10, float& w11) {
  float xf = floorf(u), yf = floorf(v);
  int x0 = (int)xf, y0 = (int)yf;
  float wx = u - xf, wy = v - yf;
  int x1 = x0 + 1, y1 = y0 + 1;
  float v00 = (x0 >= 0 && x0 < WW && y0 >= 0 && y0 < HH) ? 1.f : 0.f;
  float v01 = (x1 >= 0 && x1 < WW && y0 >= 0 && y0 < HH) ? 1.f : 0.f;
  float v10 = (x0 >= 0 && x0 < WW && y1 >= 0 && y1 < HH) ? 1.f : 0.f;
  float v11 = (x1 >= 0 && x1 < WW && y1 >= 0 && y1 < HH) ? 1.f : 0.f;
  int cx0 = x0 < 0 ? 0 : (x0 > WW - 1 ? WW - 1 : x0);
  int cx1 = x1 < 0 ? 0 : (x1 > WW - 1 ? WW - 1 : x1);
  int cy0 = y0 < 0 ? 0 : (y0 > HH - 1 ? HH - 1 : y0);
  int cy1 = y1 < 0 ? 0 : (y1 > HH - 1 ? HH - 1 : y1);
  i00 = cy0 * WW + cx0; i01 = cy0 * WW + cx1;
  i10 = cy1 * WW + cx0; i11 = cy1 * WW + cx1;
  w00 = (1.f - wx) * (1.f - wy) * v00;
  w01 = wx * (1.f - wy) * v01;
  w10 = (1.f - wx) * wy * v10;
  w11 = wx * wy * v11;
}

__device__ __forceinline__ float tap2(const uint32_t* fp, int i00, int i01,
                                      int i10, int i11, float w00, float w01,
                                      float w10, float w11, float& r1) {
  H2 t00, t01, t10, t11;
  t00.u = fp[i00]; t01.u = fp[i01]; t10.u = fp[i10]; t11.u = fp[i11];
  float r0 = w00 * (float)t00.h[0] + w01 * (float)t01.h[0] +
             w10 * (float)t10.h[0] + w11 * (float)t11.h[0];
  r1 = w00 * (float)t00.h[1] + w01 * (float)t01.h[1] +
       w10 * (float)t10.h[1] + w11 * (float)t11.h[1];
  return r0;
}

// center sample: packed feat (n,32 planes,H,W) -> packed out (n,32 planes,H,W)
__global__ void sample_center_k(const uint32_t* __restrict__ feat,
                                const float* __restrict__ dout,
                                uint32_t* __restrict__ outp) {
  int idx = blockIdx.x * blockDim.x + threadIdx.x;
  if (idx >= NB * HWP) return;
  int nb = idx / HWP, pix = idx % HWP;
  int row = pix / WW, col = pix % WW;
  float cx = -53.7f + col * 0.6f;
  float cy = -53.7f + row * 0.6f;
  float rx = dout[((size_t)(nb * 18 + 10)) * HWP + pix] + cx;
  float ry = dout[((size_t)(nb * 18 + 11)) * HWP + pix] + cy;
  int i00, i01, i10, i11; float w00, w01, w10, w11;
  bilin_setup(tgrid(rx), tgrid(ry), i00, i01, i10, i11, w00, w01, w10, w11);
  for (int cp = 0; cp < 32; ++cp) {
    const uint32_t* fp = feat + ((size_t)(nb * 32 + cp)) * HWP;
    float r1;
    float r0 = tap2(fp, i00, i01, i10, i11, w00, w01, w10, w11, r1);
    H2 o;
    o.h[0] = (_Float16)r0;
    o.h[1] = (_Float16)r1;
    outp[((size_t)(nb * 32 + cp)) * HWP + pix] = o.u;
  }
}

// 4-surface sample. mode0: template box, compute+store theta. mode1: exp(wlh).
// Output channel = hc*4 + surface; packed pairs -> plane 4*cp + {0..3}.
__global__ void sample_surf_k(const uint32_t* __restrict__ feat,
                              const float* __restrict__ dout,
                              uint32_t* __restrict__ outp,
                              float* __restrict__ thbuf, int mode) {
  int idx = blockIdx.x * blockDim.x + threadIdx.x;
  if (idx >= NB * HWP) return;
  int nb = idx / HWP, pix = idx % HWP;
  int row = pix / WW, col = pix % WW;
  float cx = -53.7f + col * 0.6f;
  float cy = -53.7f + row * 0.6f;
  float rx = dout[((size_t)(nb * 18 + 10)) * HWP + pix] + cx;
  float ry = dout[((size_t)(nb * 18 + 11)) * HWP + pix] + cy;

  float th, wbox, lbox;
  if (mode == 0) {
    float c = dout[((size_t)(nb * 18 + 16)) * HWP + pix];
    float s = dout[((size_t)(nb * 18 + 17)) * HWP + pix];
    th = atan2f(s, c);
    thbuf[(size_t)nb * HWP + pix] = th;
    wbox = 1.92f; lbox = 4.62f;
  } else {
    th = thbuf[(size_t)nb * HWP + pix];
    float w0 = dout[((size_t)(nb * 18 + 13)) * HWP + pix];  // already clipped
    float l0 = dout[((size_t)(nb * 18 + 14)) * HWP + pix];
    wbox = expf(w0); lbox = expf(l0);
    if (wbox != wbox) wbox = 0.f;
    if (lbox != lbox) lbox = 0.f;
  }
  float cth = cosf(th), sth = sinf(th);
  float hl = 0.5f * lbox, hw = 0.5f * wbox;
  float px[4], py[4];
  px[0] = rx + hl * cth; py[0] = ry + hl * sth;
  px[1] = rx - hl * cth; py[1] = ry - hl * sth;
  px[2] = rx - hw * sth; py[2] = ry + hw * cth;
  px[3] = rx + hw * sth; py[3] = ry - hw * cth;

  int i00[4], i01[4], i10[4], i11[4];
  float w00[4], w01[4], w10[4], w11[4];
#pragma unroll
  for (int s4 = 0; s4 < 4; ++s4)
    bilin_setup(tgrid(px[s4]), tgrid(py[s4]), i00[s4], i01[s4], i10[s4], i11[s4],
                w00[s4], w01[s4], w10[s4], w11[s4]);

  for (int cp = 0; cp < 32; ++cp) {  // channels c0=2cp, c1=2cp+1
    const uint32_t* fp = feat + ((size_t)(nb * 32 + cp)) * HWP;
    float r0[4], r1[4];
#pragma unroll
    for (int s4 = 0; s4 < 4; ++s4)
      r0[s4] = tap2(fp, i00[s4], i01[s4], i10[s4], i11[s4],
                    w00[s4], w01[s4], w10[s4], w11[s4], r1[s4]);
    // plane 4cp+0: (c0,s0),(c0,s1)  plane 4cp+1: (c0,s2),(c0,s3)
    // plane 4cp+2: (c1,s0),(c1,s1)  plane 4cp+3: (c1,s2),(c1,s3)
    H2 o0, o1, o2, o3;
    o0.h[0] = (_Float16)r0[0]; o0.h[1] = (_Float16)r0[1];
    o1.h[0] = (_Float16)r0[2]; o1.h[1] = (_Float16)r0[3];
    o2.h[0] = (_Float16)r1[0]; o2.h[1] = (_Float16)r1[1];
    o3.h[0] = (_Float16)r1[2]; o3.h[1] = (_Float16)r1[3];
    size_t base = ((size_t)(nb * 128 + cp * 4)) * HWP + pix;
    outp[base]           = o0.u;
    outp[base + HWP]     = o1.u;
    outp[base + 2 * HWP] = o2.u;
    outp[base + 3 * HWP] = o3.u;
  }
}

// ---------------------------------------------------------------------------
// Host orchestration
// ---------------------------------------------------------------------------
extern "C" void kernel_launch(void* const* d_in, const int* in_sizes, int n_in,
                              void* d_out, int out_size, void* d_ws, size_t ws_size,
                              hipStream_t stream) {
  const float* x         = (const float*)d_in[0];
  const float* conv1_w   = (const float*)d_in[1];
  const float* conv1_b   = (const float*)d_in[2];
  const float* xyzc_w    = (const float*)d_in[3];
  const float* xyzc_b    = (const float*)d_in[4];
  const float* xyzbn_g   = (const float*)d_in[5];
  const float* xyzbn_b   = (const float*)d_in[6];
  const float* xyzbn_m   = (const float*)d_in[7];
  const float* xyzbn_v   = (const float*)d_in[8];
  const float* xyzh_w    = (const float*)d_in[9];
  const float* xyzh_b    = (const float*)d_in[10];
  const float* thc_w     = (const float*)d_in[11];
  const float* thc_b     = (const float*)d_in[12];
  const float* thbn_g    = (const float*)d_in[13];
  const float* thbn_b    = (const float*)d_in[14];
  const float* thbn_m    = (const float*)d_in[15];
  const float* thbn_v    = (const float*)d_in[16];
  const float* thh_w     = (const float*)d_in[17];
  const float* thh_b     = (const float*)d_in[18];
  const float* wlc_w     = (const float*)d_in[19];
  const float* wlc_b     = (const float*)d_in[20];
  const float* wlbn_g    = (const float*)d_in[21];
  const float* wlbn_b    = (const float*)d_in[22];
  const float* wlbn_m    = (const float*)d_in[23];
  const float* wlbn_v    = (const float*)d_in[24];
  const float* wlh_w     = (const float*)d_in[25];
  const float* wlh_b     = (const float*)d_in[26];
  const float* clc_w     = (const float*)d_in[27];
  const float* clc_b     = (const float*)d_in[28];
  const float* clbn_g    = (const float*)d_in[29];
  const float* clbn_b    = (const float*)d_in[30];
  const float* clbn_m    = (const float*)d_in[31];
  const float* clbn_v    = (const float*)d_in[32];
  const float* clh_w     = (const float*)d_in[33];
  const float* clh_b     = (const float*)d_in[34];

  float* ws = (float*)d_ws;
  uint32_t* fbuf  = (uint32_t*)(ws);             // packed (4,32,H,W)
  uint32_t* feat  = (uint32_t*)(ws + 4147200);   // packed (4,32,H,W) reused
  uint32_t* samp  = (uint32_t*)(ws + 8294400);   // packed (4,128,H,W) reused
  float* thbuf = ws + 24883200;                  // (4,H,W)
  float* bnws  = ws + 25012800;                  // 4 layers * (s[64], t[64])
  _Float16* w16 = (_Float16*)(ws + 25013312);

  _Float16* w_conv1 = w16;
  _Float16* w_xyzc  = w16 + 147456;
  _Float16* w_xyzh  = w16 + 184320;
  _Float16* w_thc   = w16 + 186048;
  _Float16* w_thh   = w16 + 259776;
  _Float16* w_wlc   = w16 + 260928;
  _Float16* w_wlhh  = w16 + 445248;
  _Float16* w_clc   = w16 + 446976;
  _Float16* w_clh   = w16 + 631296;

  auto prep = [&](const float* src, _Float16* dst, int O, int I) {
    int total = O * I * 9;
    prep_w_k<<<(total + 255) / 256, 256, 0, stream>>>(src, dst, O, I);
  };
  prep(conv1_w, w_conv1, 64, 256);
  prep(xyzc_w,  w_xyzc,  64, 64);
  prep(xyzh_w,  w_xyzh,   3, 64);
  prep(thc_w,   w_thc,   64, 128);
  prep(thh_w,   w_thh,    2, 64);
  prep(wlc_w,   w_wlc,   64, 320);
  prep(wlh_w,   w_wlhh,   3, 64);
  prep(clc_w,   w_clc,   64, 320);
  prep(clh_w,   w_clh,   10, 64);

  prep_bn_k<<<1, 64, 0, stream>>>(xyzbn_g, xyzbn_b, xyzbn_m, xyzbn_v, bnws + 0,   bnws + 64,  64);
  prep_bn_k<<<1, 64, 0, stream>>>(thbn_g,  thbn_b,  thbn_m,  thbn_v,  bnws + 128, bnws + 192, 64);
  prep_bn_k<<<1, 64, 0, stream>>>(wlbn_g,  wlbn_b,  wlbn_m,  wlbn_v,  bnws + 256, bnws + 320, 64);
  prep_bn_k<<<1, 64, 0, stream>>>(clbn_g,  clbn_b,  clbn_m,  clbn_v,  bnws + 384, bnws + 448, 64);

  float* dout = (float*)d_out;
  const dim3 cgrid((WW + PTILE - 1) / PTILE, HH, NB);
  const int tot = NB * HWP;
  const int sgrid = (tot + 255) / 256;

  // f = relu(conv1(x)) : f32 in, packed f16 out
  conv3x3_wmma<<<cgrid, 128, 0, stream>>>(x, 256, nullptr, 0, w_conv1, conv1_b,
                                          nullptr, nullptr, fbuf, 0, 0, 64, 1, 0, 1);
  // xf = relu(bn(xyz_conv(f)))
  conv3x3_wmma<<<cgrid, 128, 0, stream>>>(fbuf, 64, nullptr, 0, w_xyzc, xyzc_b,
                                          bnws + 0, bnws + 64, feat, 0, 0, 64, 1, 1, 1);
  // xyz -> d_out ch 10..12 (f32)
  conv3x3_wmma<<<cgrid, 128, 0, stream>>>(feat, 64, nullptr, 0, w_xyzh, xyzh_b,
                                          nullptr, nullptr, dout, 18, 10, 3, 0, 1, 0);
  // xf_s = sample(xf, centers)
  sample_center_k<<<sgrid, 256, 0, stream>>>(feat, dout, samp);
  // tf = relu(bn(theta_conv(cat[f, xf_s])))
  conv3x3_wmma<<<cgrid, 128, 0, stream>>>(fbuf, 64, samp, 64, w_thc, thc_b,
                                          bnws + 128, bnws + 192, feat, 0, 0, 64, 1, 1, 1);
  // theta -> d_out ch 16..17 (f32)
  conv3x3_wmma<<<cgrid, 128, 0, stream>>>(feat, 64, nullptr, 0, w_thh, thh_b,
                                          nullptr, nullptr, dout, 18, 16, 2, 0, 1, 0);
  // tf_s = sample4(tf, template box); also computes+stores theta angle
  sample_surf_k<<<sgrid, 256, 0, stream>>>(feat, dout, samp, thbuf, 0);
  // wf = relu(bn(wlh_conv(cat[f, tf_s])))
  conv3x3_wmma<<<cgrid, 128, 0, stream>>>(fbuf, 64, samp, 256, w_wlc, wlc_b,
                                          bnws + 256, bnws + 320, feat, 0, 0, 64, 1, 1, 1);
  // wlh = clip(head(wf)) -> d_out ch 13..15 (f32)
  conv3x3_wmma<<<cgrid, 128, 0, stream>>>(feat, 64, nullptr, 0, w_wlhh, wlh_b,
                                          nullptr, nullptr, dout, 18, 13, 3, 2, 1, 0);
  // wf_s = sample4(wf, predicted box)
  sample_surf_k<<<sgrid, 256, 0, stream>>>(feat, dout, samp, thbuf, 1);
  // cf = relu(bn(cls_conv(cat[f, wf_s])))
  conv3x3_wmma<<<cgrid, 128, 0, stream>>>(fbuf, 64, samp, 256, w_clc, clc_b,
                                          bnws + 384, bnws + 448, feat, 0, 0, 64, 1, 1, 1);
  // logits -> d_out ch 0..9 (f32)
  conv3x3_wmma<<<cgrid, 128, 0, stream>>>(feat, 64, nullptr, 0, w_clh, clh_b,
                                          nullptr, nullptr, dout, 18, 0, 10, 0, 1, 0);
}
